// VAEbar8_60146722013634
// MI455X (gfx1250) — matrix operand order repaired and checked by reading, compiled, not run on hardware
//
#include <hip/hip_runtime.h>
#include <hip/hip_bf16.h>
#include <math.h>

// Dimensions from the reference
#define BB    256
#define TT    128
#define ROLLD 130
#define CONDD 12
#define RHYD  3
#define HD    1024
#define ZD    128
#define GD    3072   // 3*H (gate dim)
#define XCK   160    // ROLL+COND = 142 padded to multiple of 32

typedef __attribute__((ext_vector_type(16))) __bf16 v16bf;
typedef __attribute__((ext_vector_type(8)))  float  v8f;

static __device__ __forceinline__ unsigned short f2bf(float f) {
  union { float f; unsigned u; } c; c.f = f;
  unsigned r = 0x7FFFu + ((c.u >> 16) & 1u);          // round-to-nearest-even
  return (unsigned short)((c.u + r) >> 16);
}

// ---------------------------------------------------------------------------
// Core bf16 WMMA GEMM: C[M,N] = A[M,K] @ W[N,K]^T + bias[N]
// A,W row-major bf16 (raw u16), K multiple of 32, M multiple of 32,
// N multiple of 16.
// Register-blocked: each wave computes a 16(M) x 64(N) tile: one A fragment
// reused across 4 B fragments / 4 accumulators per k-step (4 WMMAs per
// 10 b128 loads). 8 waves/block cover a 32(M) x 256(N) block tile.
// ---------------------------------------------------------------------------
__global__ __launch_bounds__(256)
void k_wmma_gemm(const unsigned short* __restrict__ A,
                 const unsigned short* __restrict__ W,
                 const float* __restrict__ bias,
                 float* __restrict__ C,
                 int M, int N, int K)
{
  const int wave = threadIdx.x >> 5;
  const int lane = threadIdx.x & 31;
  const int hl   = lane >> 4;       // half-wave select
  const int r    = lane & 15;

  const int wm  = wave & 1;                       // 2 waves along M
  const int wn  = wave >> 1;                      // 4 waves along N
  const int tm  = (blockIdx.y * 2 + wm) * 16;
  const int tn0 = (blockIdx.x * 4 + wn) * 64;
  if (tn0 >= N) return;                           // wave-uniform exit

  const unsigned* __restrict__ Arow = (const unsigned*)(A + (size_t)(tm + r) * K);

  const unsigned* Wrow[4];
  bool valid[4];
#pragma unroll
  for (int s = 0; s < 4; ++s) {
    const int rn = tn0 + 16 * s;                  // N multiple of 16 => all-or-none
    valid[s] = (rn < N);
    const int row = valid[s] ? (rn + r) : r;      // clamp OOB subtile to row 0..15
    Wrow[s] = (const unsigned*)(W + (size_t)row * K);
  }

  v8f acc[4];
#pragma unroll
  for (int s = 0; s < 4; ++s) acc[s] = (v8f){0.f,0.f,0.f,0.f,0.f,0.f,0.f,0.f};

  for (int k = 0; k < K; k += 32) {
    union { v16bf v; unsigned u[8]; } a;
#pragma unroll
    for (int i = 0; i < 8; ++i) {
      // A 16x32 bf16 layout: VGPR i -> K pair 2*(i&3) + 8*hl + 16*(i>>2)
      const int ka = k + 2*(i & 3) + 8*hl + 16*(i >> 2);
      a.u[i] = Arow[ka >> 1];
    }
#pragma unroll
    for (int s = 0; s < 4; ++s) {
      union { v16bf v; unsigned u[8]; } b;
#pragma unroll
      for (int i = 0; i < 8; ++i) {
        // B 32x16 bf16 layout: VGPR i -> K pair 2*(i&3) + 8*hl + 16*(i>>2), column = lane%16
        const int kb = k + 2*(i & 3) + 8*hl + 16*(i >> 2);
        b.u[i] = Wrow[s][kb >> 1];
      }
      acc[s] = __builtin_amdgcn_wmma_f32_16x16x32_bf16(
          false, a.v, false, b.v, (short)0, acc[s], false, false);
    }
  }

#pragma unroll
  for (int s = 0; s < 4; ++s) {
    if (!valid[s]) continue;
    const int n = tn0 + 16 * s + r;
    const float bv = bias[n];
#pragma unroll
    for (int i = 0; i < 8; ++i) {
      const int m = tm + i + 8*hl;
      C[(size_t)m * N + n] = acc[s][i] + bv;
    }
  }
}

// ---------------------------------------------------------------------------
// fp32 -> bf16 with N/K padding (zeros): dst is Npad x Kpad, src Nsrc x Ksrc
// ---------------------------------------------------------------------------
__global__ void k_cvt_pad(const float* __restrict__ src, unsigned short* __restrict__ dst,
                          int Nsrc, int Npad, int Ksrc, int Kpad)
{
  int idx = blockIdx.x * 256 + threadIdx.x;
  if (idx >= Npad * Kpad) return;
  int n = idx / Kpad, k = idx % Kpad;
  float v = (n < Nsrc && k < Ksrc) ? src[(size_t)n * Ksrc + k] : 0.f;
  dst[idx] = f2bf(v);
}

// Pack xc = [x | condition] into bf16 (T, B, XCK) with zero pad
__global__ void k_build_xc(const float* __restrict__ x, const float* __restrict__ cond,
                           unsigned short* __restrict__ xcb)
{
  int idx = blockIdx.x * 256 + threadIdx.x;
  if (idx >= TT * BB * XCK) return;
  int t = idx / (BB * XCK);
  int rem = idx % (BB * XCK);
  int b = rem / XCK, k = rem % XCK;
  float v = 0.f;
  if (k < ROLLD)              v = x[((size_t)b * TT + t) * ROLLD + k];
  else if (k < ROLLD + CONDD) v = cond[((size_t)b * TT + t) * CONDD + (k - ROLLD)];
  xcb[idx] = f2bf(v);
}

// Fused GRU gate update: h' = (1-z)*n + z*h ; writes fp32 + bf16 shadow
__global__ void k_gru_fuse(const float* __restrict__ gi, const float* __restrict__ gh,
                           const float* __restrict__ hin,
                           float* __restrict__ hout, unsigned short* __restrict__ houtb)
{
  int idx = blockIdx.x * 256 + threadIdx.x;
  if (idx >= BB * HD) return;
  int b = idx >> 10, j = idx & (HD - 1);
  const float* gib = gi + (size_t)b * GD;
  const float* ghb = gh + (size_t)b * GD;
  float r = 1.f / (1.f + expf(-(gib[j]        + ghb[j])));
  float z = 1.f / (1.f + expf(-(gib[j + HD]   + ghb[j + HD])));
  float n = tanhf(gib[j + 2*HD] + r * ghb[j + 2*HD]);
  float h = (1.f - z) * n + z * hin[idx];
  hout[idx]  = h;
  houtb[idx] = f2bf(h);
}

// Naive fp32 projection out[b,n] = act( A[b,:K] . W[n, coff:coff+K] + bias[n] )
__global__ void k_naive_proj(const float* __restrict__ A, const float* __restrict__ W,
                             int ldw, int coff, const float* __restrict__ bias,
                             float* __restrict__ out, unsigned short* __restrict__ outb,
                             int M, int N, int K, int do_tanh)
{
  int idx = blockIdx.x * 256 + threadIdx.x;
  if (idx >= M * N) return;
  int b = idx / N, n = idx % N;
  const float* a = A + (size_t)b * K;
  const float* w = W + (size_t)n * ldw + coff;
  float s = bias[n];
  for (int k = 0; k < K; ++k) s += a[k] * w[k];
  if (do_tanh) s = tanhf(s);
  out[idx] = s;
  if (outb) outb[idx] = f2bf(s);
}

// Rhythm-decoder input proj: gi = base + Wih[:, argmax]  (one-hot gather)
__global__ void k_gi0(const float* __restrict__ base, const float* __restrict__ W0ih,
                      const int* __restrict__ amax, float* __restrict__ gi)
{
  int idx = blockIdx.x * 256 + threadIdx.x;
  if (idx >= BB * GD) return;
  int b = idx / GD, n = idx % GD;
  gi[idx] = base[idx] + W0ih[(size_t)n * (RHYD + ZD) + amax[b]];
}

// Roll-decoder gc1 input proj: one-hot gather + rhythm(3) + condition(12) terms
__global__ void k_gi1(const float* __restrict__ base, const float* __restrict__ W1ih,
                      const int* __restrict__ amax, const float* __restrict__ rhy,
                      const float* __restrict__ cond, int t, float* __restrict__ gi)
{
  int idx = blockIdx.x * 256 + threadIdx.x;
  if (idx >= BB * GD) return;
  int b = idx / GD, n = idx % GD;
  const float* w = W1ih + (size_t)n * 273;      // ROLL+RHY+Z1+COND = 273
  float s = base[idx] + w[amax[b]];
  const float* rp = rhy  + ((size_t)b * TT + t) * RHYD;
  s += rp[0] * w[ROLLD] + rp[1] * w[ROLLD + 1] + rp[2] * w[ROLLD + 2];
  const float* cp = cond + ((size_t)b * TT + t) * CONDD;
#pragma unroll
  for (int j = 0; j < CONDD; ++j) s += cp[j] * w[ROLLD + RHYD + ZD + j];
  gi[idx] = s;
}

// Rhythm head: logits(3) = hx @ Wo0^T + bo0, log-softmax -> out, argmax. Wave/b.
__global__ void k_rhy_head(const float* __restrict__ hx, const float* __restrict__ Wo0,
                           const float* __restrict__ bo0, float* __restrict__ outrhy,
                           int t, int* __restrict__ amax)
{
  int wave = threadIdx.x >> 5, lane = threadIdx.x & 31;
  int b = blockIdx.x * 8 + wave;
  const float* h = hx + (size_t)b * HD;
  float s0 = 0.f, s1 = 0.f, s2 = 0.f;
  for (int k = lane; k < HD; k += 32) {
    float hv = h[k];
    s0 += hv * Wo0[k]; s1 += hv * Wo0[HD + k]; s2 += hv * Wo0[2*HD + k];
  }
#pragma unroll
  for (int o = 16; o; o >>= 1) {
    s0 += __shfl_xor(s0, o); s1 += __shfl_xor(s1, o); s2 += __shfl_xor(s2, o);
  }
  if (lane == 0) {
    s0 += bo0[0]; s1 += bo0[1]; s2 += bo0[2];
    float m = fmaxf(s0, fmaxf(s1, s2));
    float lse = m + logf(expf(s0 - m) + expf(s1 - m) + expf(s2 - m));
    float* o = outrhy + ((size_t)b * TT + t) * RHYD;
    o[0] = s0 - lse; o[1] = s1 - lse; o[2] = s2 - lse;
    int a = 0; float bv = s0;
    if (s1 > bv) { bv = s1; a = 1; }
    if (s2 > bv) { a = 2; }
    amax[b] = a;
  }
}

// Roll head: log-softmax over first 130 of gl(256x144), write recon, argmax. Wave/b.
__global__ void k_roll_head(const float* __restrict__ gl, float* __restrict__ outroll,
                            int t, int* __restrict__ amax)
{
  int wave = threadIdx.x >> 5, lane = threadIdx.x & 31;
  int b = blockIdx.x * 8 + wave;
  const float* g = gl + (size_t)b * 144;
  float m = -3.0e38f; int mi = 0;
  for (int j = lane; j < ROLLD; j += 32) { float v = g[j]; if (v > m) { m = v; mi = j; } }
#pragma unroll
  for (int o = 16; o; o >>= 1) {
    float om = __shfl_xor(m, o); int omi = __shfl_xor(mi, o);
    if (om > m || (om == m && omi < mi)) { m = om; mi = omi; }
  }
  float s = 0.f;
  for (int j = lane; j < ROLLD; j += 32) s += expf(g[j] - m);
#pragma unroll
  for (int o = 16; o; o >>= 1) s += __shfl_xor(s, o);
  float lse = m + logf(s);
  float* o = outroll + ((size_t)b * TT + t) * ROLLD;
  for (int j = lane; j < ROLLD; j += 32) o[j] = g[j] - lse;
  if (lane == 0) amax[b] = mi;
}

// Latent: mu1/std1/mu2/std2/z1/z2 from mu,logvar buffers
__global__ void k_latent(const float* __restrict__ mu, const float* __restrict__ lv,
                         const float* __restrict__ eps1, const float* __restrict__ eps2,
                         float* __restrict__ omu1, float* __restrict__ ostd1,
                         float* __restrict__ omu2, float* __restrict__ ostd2,
                         float* __restrict__ oz1,  float* __restrict__ oz2)
{
  int idx = blockIdx.x * 256 + threadIdx.x;
  if (idx >= BB * 2 * ZD) return;
  int b = idx >> 8, n = idx & 255;
  float m = mu[idx], s = expf(lv[idx]);
  if (n < ZD) { int i = b * ZD + n;        omu1[i] = m; ostd1[i] = s; oz1[i] = m + s * eps1[i]; }
  else        { int i = b * ZD + (n - ZD); omu2[i] = m; ostd2[i] = s; oz2[i] = m + s * eps2[i]; }
}

__global__ void k_flat(const float* __restrict__ hf, const float* __restrict__ hb,
                       unsigned short* __restrict__ flatb)
{
  int idx = blockIdx.x * 256 + threadIdx.x;
  if (idx >= BB * 2 * HD) return;
  int b = idx >> 11, n = idx & 2047;
  float v = (n < HD) ? hf[(size_t)b * HD + n] : hb[(size_t)b * HD + n - HD];
  flatb[idx] = f2bf(v);
}

__global__ void k_init_h(float* __restrict__ h, unsigned short* __restrict__ hb)
{
  int idx = blockIdx.x * 256 + threadIdx.x;
  if (idx >= BB * HD) return;
  h[idx] = 0.f; hb[idx] = 0;
}

__global__ void k_init_misc(int* __restrict__ a0, int* __restrict__ a1,
                            const float* __restrict__ bo1, float* __restrict__ bo1p)
{
  int idx = blockIdx.x * 256 + threadIdx.x;
  if (idx < BB) { a0[idx] = RHYD - 1; a1[idx] = ROLLD - 1; }
  if (idx < 144) bo1p[idx] = (idx < ROLLD) ? bo1[idx] : 0.f;
}

__global__ void k_copy_h(const float* __restrict__ src, float* __restrict__ dst,
                         unsigned short* __restrict__ dstb)
{
  int idx = blockIdx.x * 256 + threadIdx.x;
  if (idx >= BB * HD) return;
  float v = src[idx]; dst[idx] = v; dstb[idx] = f2bf(v);
}

// ---------------------------------------------------------------------------
extern "C" void kernel_launch(void* const* d_in, const int* in_sizes, int n_in,
                              void* d_out, int out_size, void* d_ws, size_t ws_size,
                              hipStream_t stream) {
  (void)in_sizes; (void)n_in; (void)out_size; (void)ws_size;

  const float* x        = (const float*)d_in[0];
  const float* cond     = (const float*)d_in[1];
  const float* eps1     = (const float*)d_in[2];
  const float* eps2     = (const float*)d_in[3];
  const float* gruf_Wih = (const float*)d_in[4];
  const float* gruf_Whh = (const float*)d_in[5];
  const float* gruf_bih = (const float*)d_in[6];
  const float* gruf_bhh = (const float*)d_in[7];
  const float* grub_Wih = (const float*)d_in[8];
  const float* grub_Whh = (const float*)d_in[9];
  const float* grub_bih = (const float*)d_in[10];
  const float* grub_bhh = (const float*)d_in[11];
  const float* Wmu      = (const float*)d_in[12];
  const float* bmu      = (const float*)d_in[13];
  const float* Wvar     = (const float*)d_in[14];
  const float* bvar     = (const float*)d_in[15];
  const float* gc0_Wih  = (const float*)d_in[16];
  const float* gc0_Whh  = (const float*)d_in[17];
  const float* gc0_bih  = (const float*)d_in[18];
  const float* gc0_bhh  = (const float*)d_in[19];
  const float* gc1_Wih  = (const float*)d_in[20];
  const float* gc1_Whh  = (const float*)d_in[21];
  const float* gc1_bih  = (const float*)d_in[22];
  const float* gc1_bhh  = (const float*)d_in[23];
  const float* gc2_Wih  = (const float*)d_in[24];
  const float* gc2_Whh  = (const float*)d_in[25];
  const float* gc2_bih  = (const float*)d_in[26];
  const float* gc2_bhh  = (const float*)d_in[27];
  const float* Wi0      = (const float*)d_in[28];
  const float* bi0      = (const float*)d_in[29];
  const float* Wo0      = (const float*)d_in[30];
  const float* bo0      = (const float*)d_in[31];
  const float* Wi1      = (const float*)d_in[32];
  const float* bi1      = (const float*)d_in[33];
  const float* Wo1      = (const float*)d_in[34];
  const float* bo1      = (const float*)d_in[35];

  // d_out layout: recon | recon_rhythm | mu1 | std1 | mu2 | std2 | z1 | z2
  float* out = (float*)d_out;
  float* o_recon = out;
  float* o_rhy   = o_recon + (size_t)BB * TT * ROLLD;
  float* o_mu1   = o_rhy   + (size_t)BB * TT * RHYD;
  float* o_std1  = o_mu1   + (size_t)BB * ZD;
  float* o_mu2   = o_std1  + (size_t)BB * ZD;
  float* o_std2  = o_mu2   + (size_t)BB * ZD;
  float* o_z1    = o_std2  + (size_t)BB * ZD;
  float* o_z2    = o_z1    + (size_t)BB * ZD;

  // Workspace carve (~76 MB)
  char* base = (char*)d_ws; size_t off = 0;
  auto alloc = [&](size_t bytes) -> void* {
    off = (off + 255) & ~(size_t)255; void* p = base + off; off += bytes; return p;
  };
  unsigned short* Wf_ih  = (unsigned short*)alloc((size_t)GD * XCK * 2);
  unsigned short* Wb_ih  = (unsigned short*)alloc((size_t)GD * XCK * 2);
  unsigned short* Wf_hh  = (unsigned short*)alloc((size_t)GD * HD * 2);
  unsigned short* Wb_hh  = (unsigned short*)alloc((size_t)GD * HD * 2);
  unsigned short* W0_hh  = (unsigned short*)alloc((size_t)GD * HD * 2);
  unsigned short* W1_hh  = (unsigned short*)alloc((size_t)GD * HD * 2);
  unsigned short* W2_ih  = (unsigned short*)alloc((size_t)GD * HD * 2);
  unsigned short* W2_hh  = (unsigned short*)alloc((size_t)GD * HD * 2);
  unsigned short* Wmu_b  = (unsigned short*)alloc((size_t)256 * 2048 * 2);
  unsigned short* Wvar_b = (unsigned short*)alloc((size_t)256 * 2048 * 2);
  unsigned short* Wo1_b  = (unsigned short*)alloc((size_t)144 * HD * 2);
  unsigned short* xc_b   = (unsigned short*)alloc((size_t)TT * BB * XCK * 2);
  float* gi   = (float*)alloc((size_t)BB * GD * 4);
  float* gh   = (float*)alloc((size_t)BB * GD * 4);
  float* gl   = (float*)alloc((size_t)BB * 144 * 4);
  float* hf   = (float*)alloc((size_t)BB * HD * 4);
  float* hb   = (float*)alloc((size_t)BB * HD * 4);
  float* hx   = (float*)alloc((size_t)BB * HD * 4);
  float* h0   = (float*)alloc((size_t)BB * HD * 4);
  float* h1   = (float*)alloc((size_t)BB * HD * 4);
  unsigned short* hf_b = (unsigned short*)alloc((size_t)BB * HD * 2);
  unsigned short* hb_b = (unsigned short*)alloc((size_t)BB * HD * 2);
  unsigned short* hx_b = (unsigned short*)alloc((size_t)BB * HD * 2);
  unsigned short* h0_b = (unsigned short*)alloc((size_t)BB * HD * 2);
  unsigned short* h1_b = (unsigned short*)alloc((size_t)BB * HD * 2);
  unsigned short* flat_b = (unsigned short*)alloc((size_t)BB * 2048 * 2);
  float* mu_buf = (float*)alloc((size_t)BB * 256 * 4);
  float* lv_buf = (float*)alloc((size_t)BB * 256 * 4);
  float* giz0   = (float*)alloc((size_t)BB * GD * 4);
  float* gib1   = (float*)alloc((size_t)BB * GD * 4);
  float* bo1p   = (float*)alloc(144 * 4);
  int*   amax0  = (int*)alloc(BB * 4);
  int*   amax1  = (int*)alloc(BB * 4);

  auto cvt = [&](const float* s, unsigned short* d, int Ns, int Np, int Ks, int Kp) {
    k_cvt_pad<<<(Np * Kp + 255) / 256, 256, 0, stream>>>(s, d, Ns, Np, Ks, Kp);
  };
  auto gemm = [&](const unsigned short* A, const unsigned short* W, const float* bias,
                  float* C, int M, int N, int K) {
    dim3 g((N + 255) / 256, M / 32);
    k_wmma_gemm<<<g, 256, 0, stream>>>(A, W, bias, C, M, N, K);
  };
  const int gBH = (BB * HD) / 256;       // 1024 blocks
  const int gBG = (BB * GD) / 256;       // 3072 blocks

  // --- Stage 0: weight conversion + input packing -------------------------
  cvt(gruf_Wih, Wf_ih, GD, GD, ROLLD + CONDD, XCK);
  cvt(grub_Wih, Wb_ih, GD, GD, ROLLD + CONDD, XCK);
  cvt(gruf_Whh, Wf_hh, GD, GD, HD, HD);
  cvt(grub_Whh, Wb_hh, GD, GD, HD, HD);
  cvt(gc0_Whh,  W0_hh, GD, GD, HD, HD);
  cvt(gc1_Whh,  W1_hh, GD, GD, HD, HD);
  cvt(gc2_Wih,  W2_ih, GD, GD, HD, HD);
  cvt(gc2_Whh,  W2_hh, GD, GD, HD, HD);
  cvt(Wmu,  Wmu_b,  256, 256, 2048, 2048);
  cvt(Wvar, Wvar_b, 256, 256, 2048, 2048);
  cvt(Wo1,  Wo1_b,  ROLLD, 144, HD, HD);
  k_build_xc<<<(TT * BB * XCK + 255) / 256, 256, 0, stream>>>(x, cond, xc_b);
  k_init_h<<<gBH, 256, 0, stream>>>(hf, hf_b);
  k_init_h<<<gBH, 256, 0, stream>>>(hb, hb_b);
  k_init_misc<<<1, 256, 0, stream>>>(amax0, amax1, bo1, bo1p);

  // --- Stage 1: bidirectional GRU encoder (scan, keep final h only) -------
  for (int t = 0; t < TT; ++t) {
    // forward
    gemm(xc_b + (size_t)t * BB * XCK, Wf_ih, gruf_bih, gi, BB, GD, XCK);
    gemm(hf_b, Wf_hh, gruf_bhh, gh, BB, GD, HD);
    k_gru_fuse<<<gBH, 256, 0, stream>>>(gi, gh, hf, hf, hf_b);
    // backward (reversed time)
    gemm(xc_b + (size_t)(TT - 1 - t) * BB * XCK, Wb_ih, grub_bih, gi, BB, GD, XCK);
    gemm(hb_b, Wb_hh, grub_bhh, gh, BB, GD, HD);
    k_gru_fuse<<<gBH, 256, 0, stream>>>(gi, gh, hb, hb, hb_b);
  }

  // --- Stage 2: latent ----------------------------------------------------
  k_flat<<<(BB * 2048 + 255) / 256, 256, 0, stream>>>(hf, hb, flat_b);
  gemm(flat_b, Wmu_b,  bmu,  mu_buf, BB, 256, 2048);
  gemm(flat_b, Wvar_b, bvar, lv_buf, BB, 256, 2048);
  k_latent<<<(BB * 256 + 255) / 256, 256, 0, stream>>>(
      mu_buf, lv_buf, eps1, eps2, o_mu1, o_std1, o_mu2, o_std2, o_z1, o_z2);

  // Decoder initial states + z-dependent input-projection bases
  k_naive_proj<<<gBH, 256, 0, stream>>>(o_z2, Wi0, ZD, 0, bi0, hx, hx_b, BB, HD, ZD, 1);
  k_naive_proj<<<gBH, 256, 0, stream>>>(o_z1, Wi1, ZD, 0, bi1, h0, h0_b, BB, HD, ZD, 1);
  k_naive_proj<<<gBG, 256, 0, stream>>>(o_z2, gc0_Wih, RHYD + ZD, RHYD, gc0_bih,
                                        giz0, (unsigned short*)nullptr, BB, GD, ZD, 0);
  k_naive_proj<<<gBG, 256, 0, stream>>>(o_z1, gc1_Wih, 273, ROLLD + RHYD, gc1_bih,
                                        gib1, (unsigned short*)nullptr, BB, GD, ZD, 0);

  // --- Stage 3: rhythm decoder (sequential, argmax feedback) --------------
  for (int t = 0; t < TT; ++t) {
    k_gi0<<<gBG, 256, 0, stream>>>(giz0, gc0_Wih, amax0, gi);
    gemm(hx_b, W0_hh, gc0_bhh, gh, BB, GD, HD);
    k_gru_fuse<<<gBH, 256, 0, stream>>>(gi, gh, hx, hx, hx_b);
    k_rhy_head<<<BB / 8, 256, 0, stream>>>(hx, Wo0, bo0, o_rhy, t, amax0);
  }

  // --- Stage 4: roll decoder (2-layer GRU, sequential) --------------------
  for (int t = 0; t < TT; ++t) {
    k_gi1<<<gBG, 256, 0, stream>>>(gib1, gc1_Wih, amax1, o_rhy, cond, t, gi);
    gemm(h0_b, W1_hh, gc1_bhh, gh, BB, GD, HD);
    k_gru_fuse<<<gBH, 256, 0, stream>>>(gi, gh, h0, h0, h0_b);
    if (t == 0)
      k_copy_h<<<gBH, 256, 0, stream>>>(h0, h1, h1_b);   // h1_ = h0_ at i==0
    gemm(h0_b, W2_ih, gc2_bih, gi, BB, GD, HD);
    gemm(h1_b, W2_hh, gc2_bhh, gh, BB, GD, HD);
    k_gru_fuse<<<gBH, 256, 0, stream>>>(gi, gh, h1, h1, h1_b);
    gemm(h1_b, Wo1_b, bo1p, gl, BB, 144, HD);
    k_roll_head<<<BB / 8, 256, 0, stream>>>(gl, o_recon, t, amax1);
  }
}
